// RNN_20804821582405
// MI455X (gfx1250) — compile-verified, matrix operand order checked by reading, not run on hardware
//
#include <hip/hip_runtime.h>
#include <stdint.h>

// ---------------------------------------------------------------------------
// RNN scan on MI455X (gfx1250, wave32, WMMA).
//   h_{t+1} = relu(x[:,t]*w_in + b_in + h_t @ W)   (B=128, H=2048, T=256)
//   out     = h_T @ W_out + b_out                  (O=128)
//
// ~275 GFLOP, weights (8MB bf16) L2-resident -> compute bound. Step GEMM:
// per-WG tile M=64 x N=64 (4 waves), K in 64-wide chunks staged through LDS
// with a TRIPLE-buffered async global->LDS pipeline: chunk k+2's copies are
// issued while chunk k computes; s_wait_asynccnt 8 (partial, in-order
// completion) retires chunk k+1 without draining k+2. Grid = 64 WGs/step.
// ---------------------------------------------------------------------------

typedef __attribute__((ext_vector_type(16))) __bf16        v16bf;
typedef __attribute__((ext_vector_type(8)))  float         v8f;
typedef __attribute__((ext_vector_type(4)))  unsigned int  v4u;
typedef __attribute__((ext_vector_type(4)))  int           v4i;

union FragAB { v16bf v; v4u q[2]; };

#define KC      64            // K elements staged per pipeline stage
#define LDS_ROW 72            // 64 bf16 payload + 8 pad -> 144B row (16B align)
#define NSTAGE  3

__device__ __forceinline__ unsigned short f2bf(float f) {
  unsigned int u = __float_as_uint(f);
  u += 0x7fffu + ((u >> 16) & 1u);   // round-to-nearest-even
  return (unsigned short)(u >> 16);
}

// One 16-byte lane-chunk: global -> LDS, tracked by ASYNCcnt.
__device__ __forceinline__ void async_copy16(const unsigned short* g,
                                             unsigned short* l) {
#if __has_builtin(__builtin_amdgcn_global_load_async_to_lds_b128)
  __builtin_amdgcn_global_load_async_to_lds_b128(
      (__attribute__((address_space(1))) v4i*)(v4i*)(void*)g,
      (__attribute__((address_space(3))) v4i*)(v4i*)(void*)l,
      0, 0);
#else
  *(v4u*)l = *(const v4u*)g;   // sync fallback: global_load + ds_store
#endif
}

__device__ __forceinline__ void wait_async0() {
#if __has_builtin(__builtin_amdgcn_s_wait_asynccnt)
  __builtin_amdgcn_s_wait_asynccnt(0);
#else
  asm volatile("s_wait_asynccnt 0x0" ::: "memory");
#endif
}

// Partial wait: allow the newest stage (8 async ops) to remain in flight.
__device__ __forceinline__ void wait_async8() {
#if __has_builtin(__builtin_amdgcn_s_wait_asynccnt)
  __builtin_amdgcn_s_wait_asynccnt(8);
#else
  asm volatile("s_wait_asynccnt 0x8" ::: "memory");
#endif
}

__device__ __forceinline__ void wg_barrier() {
  asm volatile("" ::: "memory");
#if __has_builtin(__builtin_amdgcn_s_barrier_signal) && \
    __has_builtin(__builtin_amdgcn_s_barrier_wait)
  __builtin_amdgcn_s_barrier_signal(-1);
  __builtin_amdgcn_s_barrier_wait(-1);
#else
  __syncthreads();
#endif
  asm volatile("" ::: "memory");
}

// Stage one K-chunk: A tile 64xKC from gA (row base applied by caller),
// B tile 64xKC from gB rows [nbase, nbase+64). 128 threads, 8 copies each.
__device__ __forceinline__ void stage_tile(const unsigned short* __restrict__ gA,
                                           const unsigned short* __restrict__ gB,
                                           unsigned short* sA, unsigned short* sB,
                                           int k0, int H, int tid) {
#pragma unroll
  for (int i = 0; i < 4; ++i) {          // A: 512 16B-chunks / 128 threads
    const int c = tid + i * 128;
    const int row = c >> 3, kc = c & 7;
    async_copy16(gA + (long long)row * H + k0 + kc * 8,
                 sA + row * LDS_ROW + kc * 8);
  }
#pragma unroll
  for (int i = 0; i < 4; ++i) {          // B: 512 16B-chunks / 128 threads
    const int c = tid + i * 128;
    const int row = c >> 3, kc = c & 7;
    async_copy16(gB + (long long)row * H + k0 + kc * 8,
                 sB + row * LDS_ROW + kc * 8);
  }
}

// dst[c*R + r] = bf16(src[r*C + c])   (src is [R][C] row-major; R = K dim)
__global__ void rnn_cvt_transpose_bf16(const float* __restrict__ src,
                                       unsigned short* __restrict__ dst,
                                       int R, int C, long long total) {
  long long idx = (long long)blockIdx.x * blockDim.x + threadIdx.x;
  if (idx >= total) return;
  int r = (int)(idx % R);
  int c = (int)(idx / R);
  dst[(long long)c * R + r] = f2bf(src[(long long)r * C + c]);
}

// h after timestep 0: relu(x[b,0]*w_in + b_in)   (h starts at zero)
__global__ void rnn_init_h(const float* __restrict__ x,
                           const float* __restrict__ iw,
                           const float* __restrict__ ib,
                           unsigned short* __restrict__ h,
                           int T, int H, int total) {
  int idx = blockIdx.x * blockDim.x + threadIdx.x;
  if (idx >= total) return;
  int b = idx / H, n = idx % H;
  float v = x[(long long)b * T] * iw[n] + ib[n];
  h[idx] = f2bf(v > 0.f ? v : 0.f);
}

// One recurrent step: hnext = relu(x[:,t]*iw + ib + hcur @ W)
// A = hcur [128][H] bf16; B = Wt [H(n)][H(k)] bf16 (W transposed).
// 128 threads = 4 waves; wave w -> 16 M rows; block -> 64M x 64N tile.
__global__ __launch_bounds__(128) void rnn_step(
    const unsigned short* __restrict__ hcur,
    const unsigned short* __restrict__ Wt,
    const float* __restrict__ x,
    const float* __restrict__ iw,
    const float* __restrict__ ib,
    unsigned short* __restrict__ hnext,
    int t, int T, int H) {
  __shared__ __align__(16) unsigned short sA[NSTAGE][64 * LDS_ROW];
  __shared__ __align__(16) unsigned short sB[NSTAGE][64 * LDS_ROW];

  const int tid    = threadIdx.x;
  const int wave   = tid >> 5;
  const int lane   = tid & 31;
  const int m      = lane & 15;     // row (A) / col (B) within 16-subtile
  const int ks     = lane >> 4;     // K-half select per ISA bf16 frag layout
  const int mtbase = blockIdx.y * 64;
  const int nbase  = blockIdx.x * 64;
  const int mbase  = wave * 16;     // within the 64-row M tile

  const unsigned short* gA = hcur + (long long)mtbase * H;
  const unsigned short* gB = Wt + (long long)nbase * H;

  v8f acc[4] = {v8f{}, v8f{}, v8f{}, v8f{}};

  stage_tile(gA, gB, sA[0], sB[0], 0, H, tid);
  stage_tile(gA, gB, sA[1], sB[1], KC, H, tid);
  wait_async8();     // stage 0 landed (in-order), stage 1 may be in flight
  wg_barrier();

  const int nch = H / KC;
  int cur = 0;
  for (int kc = 0; kc < nch; ++kc) {
    const bool more = (kc + 2) < nch;
    if (more) {
      int nxt = cur + 2; if (nxt >= NSTAGE) nxt -= NSTAGE;
      stage_tile(gA, gB, sA[nxt], sB[nxt], (kc + 2) * KC, H, tid);
    }

    // Two 32-deep sub-chunks of this stage's 64-deep tile.
#pragma unroll
    for (int kk = 0; kk < 2; ++kk) {
      const int kofs = kk * 32 + ks * 8;
      FragAB a, b[4];
      const unsigned short* aR = &sA[cur][(mbase + m) * LDS_ROW + kofs];
      a.q[0] = *(const v4u*)(aR);          // ds_load_b128
      a.q[1] = *(const v4u*)(aR + 16);
#pragma unroll
      for (int j = 0; j < 4; ++j) {
        const unsigned short* bR = &sB[cur][(16 * j + m) * LDS_ROW + kofs];
        b[j].q[0] = *(const v4u*)(bR);
        b[j].q[1] = *(const v4u*)(bR + 16);
      }
#pragma unroll
      for (int j = 0; j < 4; ++j)
        acc[j] = __builtin_amdgcn_wmma_f32_16x16x32_bf16(
            false, a.v, false, b[j].v, (short)0, acc[j], false, false);
    }

    if (more) wait_async8();   // chunk kc+1 landed; kc+2 still in flight
    else      wait_async0();   // pipeline tail
    wg_barrier();

    if (++cur == NSTAGE) cur = 0;
  }

  // C/D layout: VGPR r, lanes 0-15 -> M=r, lanes 16-31 -> M=r+8; N = lane&15.
#pragma unroll
  for (int j = 0; j < 4; ++j) {
    const int col = nbase + 16 * j + m;
    const float wi = iw[col];
    const float bi = ib[col];
#pragma unroll
    for (int r = 0; r < 8; ++r) {
      const int row = mtbase + mbase + r + 8 * ks;
      float v = acc[j][r] + x[(long long)row * T + t] * wi + bi;
      v = v > 0.f ? v : 0.f;
      hnext[(long long)row * H + col] = f2bf(v);
    }
  }
}

// out = h_final @ W_out + b_out, f32 result. Ot is [O(n)][H(k)] bf16.
__global__ __launch_bounds__(256) void rnn_out_gemm(
    const unsigned short* __restrict__ hfin,
    const unsigned short* __restrict__ Ot,
    const float* __restrict__ bo,
    float* __restrict__ out,
    int H, int O) {
  const int wave  = threadIdx.x >> 5;
  const int lane  = threadIdx.x & 31;
  const int m     = lane & 15;
  const int ks    = lane >> 4;
  const int mbase = wave * 16;
  const int nbase = blockIdx.x * 64;
  const int ko    = ks * 8;

  const unsigned short* arow = hfin + (long long)(mbase + m) * H;
  const unsigned short* brow[4];
#pragma unroll
  for (int j = 0; j < 4; ++j)
    brow[j] = Ot + (long long)(nbase + 16 * j + m) * H;

  v8f acc[4] = {v8f{}, v8f{}, v8f{}, v8f{}};

  for (int k0 = 0; k0 < H; k0 += 32) {
    FragAB a;
    a.q[0] = *(const v4u*)(arow + k0 + ko);
    a.q[1] = *(const v4u*)(arow + k0 + ko + 16);
    __builtin_prefetch(arow + k0 + 128, 0, 3);   // global_prefetch_b8
#pragma unroll
    for (int j = 0; j < 4; ++j) {
      FragAB b;
      b.q[0] = *(const v4u*)(brow[j] + k0 + ko);
      b.q[1] = *(const v4u*)(brow[j] + k0 + ko + 16);
      acc[j] = __builtin_amdgcn_wmma_f32_16x16x32_bf16(
          false, a.v, false, b.v, (short)0, acc[j], false, false);
    }
  }

#pragma unroll
  for (int j = 0; j < 4; ++j) {
    const int col = nbase + 16 * j + m;
    const float bb = bo[col];
#pragma unroll
    for (int r = 0; r < 8; ++r) {
      const int row = mbase + r + 8 * ks;
      out[(long long)row * O + col] = acc[j][r] + bb;
    }
  }
}

extern "C" void kernel_launch(void* const* d_in, const int* in_sizes, int n_in,
                              void* d_out, int out_size, void* d_ws, size_t ws_size,
                              hipStream_t stream) {
  const float* x  = (const float*)d_in[0];   // [B,T]
  const float* iw = (const float*)d_in[1];   // [1,H]
  const float* ib = (const float*)d_in[2];   // [H]
  const float* W  = (const float*)d_in[3];   // [H,H]
  const float* Wo = (const float*)d_in[4];   // [H,O]
  const float* bo = (const float*)d_in[5];   // [O]
  float* out = (float*)d_out;                // [B,O] f32

  const int H = in_sizes[1];
  const int O = in_sizes[5];
  const int B = out_size / O;
  const int T = in_sizes[0] / B;

  char* ws = (char*)d_ws;
  unsigned short* Wt = (unsigned short*)ws;                 // [H][H] bf16 (W^T)
  size_t szW = (size_t)H * H * 2;
  unsigned short* Ot = (unsigned short*)(ws + szW);         // [O][H] bf16 (Wo^T)
  size_t szO = (size_t)H * O * 2;
  unsigned short* hA = (unsigned short*)(ws + szW + szO);   // [B][H] bf16
  size_t szh = (size_t)B * H * 2;
  unsigned short* hB = (unsigned short*)(ws + szW + szO + szh);

  const long long totW = (long long)H * H;
  const long long totO = (long long)H * O;
  const int totH = B * H;

  rnn_cvt_transpose_bf16<<<(unsigned)((totW + 255) / 256), 256, 0, stream>>>(W, Wt, H, H, totW);
  rnn_cvt_transpose_bf16<<<(unsigned)((totO + 255) / 256), 256, 0, stream>>>(Wo, Ot, H, O, totO);
  rnn_init_h<<<(totH + 255) / 256, 256, 0, stream>>>(x, iw, ib, hA, T, H, totH);

  unsigned short* cur = hA;
  unsigned short* nxt = hB;
  const dim3 sgrid(H / 64, B / 64);
  for (int t = 1; t < T; ++t) {
    rnn_step<<<sgrid, 128, 0, stream>>>(cur, Wt, x, iw, ib, nxt, t, T, H);
    unsigned short* tmp = cur; cur = nxt; nxt = tmp;
  }

  rnn_out_gemm<<<O / 64, 256, 0, stream>>>(cur, Ot, bo, out, H, O);
}
